// EntityAttentionLayer_39728447488185
// MI455X (gfx1250) — compile-verified, matrix operand order checked by reading, not run on hardware
//
#include <hip/hip_runtime.h>

typedef _Float16 half16  __attribute__((ext_vector_type(16)));
typedef float    float8  __attribute__((ext_vector_type(8)));
typedef float    floatx2 __attribute__((ext_vector_type(2)));

namespace {

constexpr int kBS = 1024, kNE = 128, kInDim = 512, kEmbed = 512, kOutDim = 512;
constexpr int kHeads = 8, kAgents = 32, kHeadDim = 64;

__device__ inline float8 wmma_f16f32(half16 a, half16 b, float8 c) {
  // v_wmma_f32_16x16x32_f16 : D = A(16x32 f16) * B(32x16 f16) + C(f32)
  return __builtin_amdgcn_wmma_f32_16x16x32_f16(false, a, false, b, (short)0, c,
                                                false, false);
}
__device__ inline float8 wmma_f32x4(floatx2 a, floatx2 b, float8 c) {
  // v_wmma_f32_16x16x4_f32 : D = A(16x4 f32) * B(4x16 f32) + C(f32)
  return __builtin_amdgcn_wmma_f32_16x16x4_f32(false, a, false, b, (short)0, c,
                                               false, false);
}

// 32-bit LDS byte offset of a __shared__ object (for async-to-LDS dest VGPR).
__device__ inline unsigned lds_offset(const void* p) {
  return (unsigned)(size_t)(__attribute__((address_space(3))) const void*)p;
}
// Async DMA: copy 16 bytes global -> LDS, tracked by ASYNCcnt (CDNA5).
__device__ inline void async_b128(const void* g, unsigned lds_off) {
  asm volatile("global_load_async_to_lds_b128 %0, %1, off"
               :
               : "v"(lds_off), "v"(g)
               : "memory");
}
__device__ inline void wait_async0() {
  asm volatile("s_wait_asynccnt 0x0" ::: "memory");
}

// ---------------------------------------------------------------------------
// Kernel P: split f32 -> (hi, lo) f16 pair images (x = hi + lo exactly to
// ~f32 precision).  Enables pure-DMA staging + 3-WMMA split-f16 GEMM.
// ---------------------------------------------------------------------------
__global__ __launch_bounds__(256) void split_f16_kernel(
    const float* __restrict__ src, _Float16* __restrict__ hi,
    _Float16* __restrict__ lo, int n4) {
  const int i = blockIdx.x * blockDim.x + threadIdx.x;
  if (i >= n4) return;
  const float4 v = ((const float4*)src)[i];
  union U { _Float16 h[4]; uint2 u; };
  U H, L;
  const float* f = (const float*)&v;
#pragma unroll
  for (int j = 0; j < 4; ++j) {
    const float x = f[j];
    const _Float16 xh = (_Float16)x;
    H.h[j] = xh;
    L.h[j] = (_Float16)(x - (float)xh);
  }
  ((uint2*)hi)[i] = H.u;
  ((uint2*)lo)[i] = L.u;
}

// ---------------------------------------------------------------------------
// Kernel A: Y[m, n] = Arow(m) . W[n]  (K = 512), split-f16 WMMA with f32 acc.
// A/B provided as pre-split f16 hi/lo images. Tile 128x128x32, 8 waves,
// each wave 32x64.  Async-to-LDS double-buffered staging (1 barrier/iter).
// grid = (N/128, M/128): consecutive blocks share the A tile -> L2 locality.
// RPB = entity rows per batch used (128 => K/V GEMM, 32 => agent-row Q GEMM).
// Columns >= 512 go to out1 (V), below to out0 (K or Q).
// ---------------------------------------------------------------------------
template <int RPB>
__global__ __launch_bounds__(256) void qkv_gemm_kernel(
    const _Float16* __restrict__ Ahi, const _Float16* __restrict__ Alo,
    const _Float16* __restrict__ Whi, const _Float16* __restrict__ Wlo,
    float* __restrict__ out0, float* __restrict__ out1) {
  // [2] double buffers; rows padded to 40 halves (80B) for bank spread.
  __shared__ alignas(16) _Float16 sAh[2][128 * 40];
  __shared__ alignas(16) _Float16 sAl[2][128 * 40];
  __shared__ alignas(16) _Float16 sBh[2][128 * 40];
  __shared__ alignas(16) _Float16 sBl[2][128 * 40];

  const int tid = threadIdx.x;
  const int wave = tid >> 5, lane = tid & 31;
  const int lm = lane & 15;
  const int n0 = blockIdx.x * 128;
  const int m0 = blockIdx.y * 128;
  const int wm = (wave & 3) * 32;   // 4 waves along M
  const int wn = (wave >> 2) * 64;  // 2 waves along N

  float8 acc[2][4];
#pragma unroll
  for (int i = 0; i < 2; ++i)
#pragma unroll
    for (int j = 0; j < 4; ++j)
#pragma unroll
      for (int r = 0; r < 8; ++r) acc[i][j][r] = 0.f;

  // staging map: each thread owns one (row, 16-half chunk) per array
  const int srow = tid >> 1;
  const int sko = (tid & 1) << 4;
  const int am = m0 + srow;
  const size_t aoff =
      ((size_t)(am / RPB) * kNE + (am % RPB)) * (size_t)kInDim + sko;
  const size_t boff = (size_t)(n0 + srow) * (size_t)kInDim + sko;
  const int sdst = srow * 40 + sko;

  // prologue: stage k=0 into buffer 0
  {
    const unsigned la = lds_offset(&sAh[0][sdst]);
    const unsigned lb = lds_offset(&sBh[0][sdst]);
    const unsigned lc = lds_offset(&sAl[0][sdst]);
    const unsigned ld = lds_offset(&sBl[0][sdst]);
    async_b128(Ahi + aoff, la);      async_b128(Ahi + aoff + 8, la + 16);
    async_b128(Alo + aoff, lc);      async_b128(Alo + aoff + 8, lc + 16);
    async_b128(Whi + boff, lb);      async_b128(Whi + boff + 8, lb + 16);
    async_b128(Wlo + boff, ld);      async_b128(Wlo + boff + 8, ld + 16);
  }

  for (int k0 = 0, it = 0; k0 < kInDim; k0 += 32, ++it) {
    const int cur = it & 1;
    wait_async0();    // my async copies into buf[cur] have landed in LDS
    __syncthreads();  // everyone's tile k is ready; prev reads of buf[cur^1] done

    if (k0 + 32 < kInDim) {  // issue DMA for tile k+1 into the other buffer
      const int nxt = cur ^ 1;
      const size_t ka = aoff + k0 + 32, kb = boff + k0 + 32;
      const unsigned la = lds_offset(&sAh[nxt][sdst]);
      const unsigned lb = lds_offset(&sBh[nxt][sdst]);
      const unsigned lc = lds_offset(&sAl[nxt][sdst]);
      const unsigned ld = lds_offset(&sBl[nxt][sdst]);
      async_b128(Ahi + ka, la);      async_b128(Ahi + ka + 8, la + 16);
      async_b128(Alo + ka, lc);      async_b128(Alo + ka + 8, lc + 16);
      async_b128(Whi + kb, lb);      async_b128(Whi + kb + 8, lb + 16);
      async_b128(Wlo + kb, ld);      async_b128(Wlo + kb + 8, ld + 16);
    }

    // fragment loads per ISA 16-bit A layout:
    // lane<16 -> K 0..7,16..23 ; lane>=16 -> K 8..15,24..31
    const int kh = (lane >> 4) << 3;
    half16 ahf[2], alf[2], bhf[4], blf[4];
#pragma unroll
    for (int mi = 0; mi < 2; ++mi) {
      const int base = (wm + mi * 16 + lm) * 40 + kh;
      union { half16 v; uint4 q[2]; } u;
      u.q[0] = *(const uint4*)&sAh[cur][base];
      u.q[1] = *(const uint4*)&sAh[cur][base + 16];
      ahf[mi] = u.v;
      u.q[0] = *(const uint4*)&sAl[cur][base];
      u.q[1] = *(const uint4*)&sAl[cur][base + 16];
      alf[mi] = u.v;
    }
#pragma unroll
    for (int ni = 0; ni < 4; ++ni) {
      const int base = (wn + ni * 16 + lm) * 40 + kh;
      union { half16 v; uint4 q[2]; } u;
      u.q[0] = *(const uint4*)&sBh[cur][base];
      u.q[1] = *(const uint4*)&sBh[cur][base + 16];
      bhf[ni] = u.v;
      u.q[0] = *(const uint4*)&sBl[cur][base];
      u.q[1] = *(const uint4*)&sBl[cur][base + 16];
      blf[ni] = u.v;
    }
#pragma unroll
    for (int mi = 0; mi < 2; ++mi)
#pragma unroll
      for (int ni = 0; ni < 4; ++ni) {
        acc[mi][ni] = wmma_f16f32(ahf[mi], bhf[ni], acc[mi][ni]);  // hi*hi
        acc[mi][ni] = wmma_f16f32(ahf[mi], blf[ni], acc[mi][ni]);  // hi*lo
        acc[mi][ni] = wmma_f16f32(alf[mi], bhf[ni], acc[mi][ni]);  // lo*hi
      }
  }

  const int hi8 = (lane >> 4) << 3;
#pragma unroll
  for (int mi = 0; mi < 2; ++mi)
#pragma unroll
    for (int ni = 0; ni < 4; ++ni) {
      const int gn = n0 + wn + ni * 16 + lm;
      float* dst = out0;
      int cn = gn;
      if (gn >= kEmbed) { dst = out1; cn = gn - kEmbed; }
#pragma unroll
      for (int r = 0; r < 8; ++r) {
        const int gm = m0 + wm + mi * 16 + r + hi8;
        dst[(size_t)gm * kEmbed + cn] = acc[mi][ni][r];
      }
    }
}

// ---------------------------------------------------------------------------
// Kernel B: per-batch fused attention + output projection.  One WG per batch,
// wave == head.  Exact f32 WMMA (16x16x4).
// ---------------------------------------------------------------------------
__global__ __launch_bounds__(256) void attn_out_kernel(
    const float* __restrict__ Q, const float* __restrict__ K,
    const float* __restrict__ V, const unsigned char* __restrict__ pre_mask,
    const unsigned char* __restrict__ post_mask,
    const float* __restrict__ Wout, const float* __restrict__ bout,
    float* __restrict__ out) {
  // per-head 32x132 f32 region: logits -> softmax weights -> (reused, stride 66) attn
  __shared__ alignas(16) float sL[kHeads * 32 * 132];

  const int b = blockIdx.x;
  const int tid = threadIdx.x;
  const int h = tid >> 5;  // wave index == head
  const int lane = tid & 31;
  const int lm = lane & 15;
  const int hi2 = (lane >> 4) << 1;  // f32-WMMA K offset: 0 or 2
  const int hi8 = (lane >> 4) << 3;  // C-matrix M offset: 0 or 8

  float* myL = sL + h * (32 * 132);
  const float* Qb = Q + (size_t)b * kAgents * kEmbed + h * kHeadDim;
  const float* Kb = K + (size_t)b * kNE * kEmbed + h * kHeadDim;
  const float* Vb = V + (size_t)b * kNE * kEmbed + h * kHeadDim;

  // ---- logits = Q_h K_h^T  (M=32, N=128, K=64) ----
  {
    float8 c[2][8];
#pragma unroll
    for (int i = 0; i < 2; ++i)
#pragma unroll
      for (int j = 0; j < 8; ++j)
#pragma unroll
        for (int r = 0; r < 8; ++r) c[i][j][r] = 0.f;

    for (int k0 = 0; k0 < kHeadDim; k0 += 4) {
      const int kk = k0 + hi2;
      floatx2 af[2], bf[8];
#pragma unroll
      for (int mi = 0; mi < 2; ++mi)
        af[mi] = *(const floatx2*)(Qb + (size_t)(mi * 16 + lm) * kEmbed + kk);
#pragma unroll
      for (int ni = 0; ni < 8; ++ni)
        bf[ni] = *(const floatx2*)(Kb + (size_t)(ni * 16 + lm) * kEmbed + kk);
#pragma unroll
      for (int mi = 0; mi < 2; ++mi)
#pragma unroll
        for (int ni = 0; ni < 8; ++ni)
          c[mi][ni] = wmma_f32x4(af[mi], bf[ni], c[mi][ni]);
    }
#pragma unroll
    for (int mi = 0; mi < 2; ++mi)
#pragma unroll
      for (int ni = 0; ni < 8; ++ni)
#pragma unroll
        for (int r = 0; r < 8; ++r)
          myL[(mi * 16 + r + hi8) * 132 + ni * 16 + lm] = c[mi][ni][r];
  }

  // ---- masked softmax: one agent row per lane (32 lanes == 32 agents) ----
  {
    const unsigned char* pm = pre_mask + ((size_t)b * kAgents + lane) * kNE;
    float* row = myL + lane * 132;
    float mx = -__builtin_inff();
    for (int e = 0; e < kNE; ++e)
      if (!pm[e]) mx = fmaxf(mx, row[e] * 0.125f);
    if (mx > -__builtin_inff()) {
      float sum = 0.f;
      for (int e = 0; e < kNE; ++e) {
        const float v = pm[e] ? 0.f : __expf(row[e] * 0.125f - mx);
        row[e] = v;
        sum += v;
      }
      const float inv = 1.f / sum;
      for (int e = 0; e < kNE; ++e) row[e] *= inv;
    } else {  // fully masked row: softmax -> NaN -> 0 per reference
      for (int e = 0; e < kNE; ++e) row[e] = 0.f;
    }
  }

  // ---- attn_h = w @ V_h  (M=32, N=64, K=128) ----
  {
    float8 d[2][4];
#pragma unroll
    for (int i = 0; i < 2; ++i)
#pragma unroll
      for (int j = 0; j < 4; ++j)
#pragma unroll
        for (int r = 0; r < 8; ++r) d[i][j][r] = 0.f;

    for (int e0 = 0; e0 < kNE; e0 += 4) {
      const int ke = e0 + hi2;
      floatx2 af[2], bf[4];
#pragma unroll
      for (int mi = 0; mi < 2; ++mi)
        af[mi] = *(const floatx2*)(myL + (size_t)(mi * 16 + lm) * 132 + ke);
#pragma unroll
      for (int ni = 0; ni < 4; ++ni) {
        const int dc = ni * 16 + lm;
        floatx2 t;
        t[0] = Vb[(size_t)ke * kEmbed + dc];
        t[1] = Vb[(size_t)(ke + 1) * kEmbed + dc];
        bf[ni] = t;
      }
#pragma unroll
      for (int mi = 0; mi < 2; ++mi)
#pragma unroll
        for (int ni = 0; ni < 4; ++ni)
          d[mi][ni] = wmma_f32x4(af[mi], bf[ni], d[mi][ni]);
    }
    float* myA = sL + h * (32 * 132);  // reuse (w fully consumed, same wave)
#pragma unroll
    for (int mi = 0; mi < 2; ++mi)
#pragma unroll
      for (int ni = 0; ni < 4; ++ni)
#pragma unroll
        for (int r = 0; r < 8; ++r)
          myA[(mi * 16 + r + hi8) * 66 + ni * 16 + lm] = d[mi][ni][r];
  }
  __syncthreads();

  // ---- out = attn @ Wout^T + b  (M=32, per-wave N=64, K=512) ----
  {
    const int nb = h * 64;
    float8 o[2][4];
#pragma unroll
    for (int i = 0; i < 2; ++i)
#pragma unroll
      for (int j = 0; j < 4; ++j)
#pragma unroll
        for (int r = 0; r < 8; ++r) o[i][j][r] = 0.f;

    for (int k0 = 0; k0 < kEmbed; k0 += 4) {
      const int kk = k0 + hi2;
      const int hh = kk >> 6;   // which head's attn region
      const int dd = kk & 63;   // offset within head (pair never crosses 64)
      const float* ab = sL + hh * (32 * 132);
      floatx2 af[2], bf[4];
#pragma unroll
      for (int mi = 0; mi < 2; ++mi)
        af[mi] = *(const floatx2*)(ab + (size_t)(mi * 16 + lm) * 66 + dd);
#pragma unroll
      for (int ni = 0; ni < 4; ++ni)
        bf[ni] =
            *(const floatx2*)(Wout + (size_t)(nb + ni * 16 + lm) * kEmbed + kk);
#pragma unroll
      for (int mi = 0; mi < 2; ++mi)
#pragma unroll
        for (int ni = 0; ni < 4; ++ni)
          o[mi][ni] = wmma_f32x4(af[mi], bf[ni], o[mi][ni]);
    }
#pragma unroll
    for (int mi = 0; mi < 2; ++mi)
#pragma unroll
      for (int ni = 0; ni < 4; ++ni) {
        const int gn = nb + ni * 16 + lm;
        const float bo = bout[gn];
#pragma unroll
        for (int r = 0; r < 8; ++r) {
          const int a = mi * 16 + r + hi8;
          float v = o[mi][ni][r] + bo;
          if (post_mask[(size_t)b * kAgents + a]) v = 0.f;
          out[((size_t)b * kAgents + a) * kOutDim + gn] = v;
        }
      }
  }
}

}  // namespace

extern "C" void kernel_launch(void* const* d_in, const int* in_sizes, int n_in,
                              void* d_out, int out_size, void* d_ws,
                              size_t ws_size, hipStream_t stream) {
  (void)in_sizes; (void)n_in; (void)out_size; (void)ws_size;
  const float* entities = (const float*)d_in[0];
  const unsigned char* pre_mask = (const unsigned char*)d_in[1];   // bool
  const unsigned char* post_mask = (const unsigned char*)d_in[2];  // bool
  const float* W_in = (const float*)d_in[3];
  const float* W_out = (const float*)d_in[4];
  const float* b_out = (const float*)d_in[5];
  float* out = (float*)d_out;

  // workspace layout (f32 then f16 images):
  //   Q  : BS*32*512  f32   ( 64 MB)
  //   K  : BS*128*512 f32   (256 MB)
  //   V  : BS*128*512 f32   (256 MB)
  //   Ehi/Elo : BS*128*512 f16 each (128 MB each)
  //   Whi/Wlo : 1536*512   f16 each (1.5 MB each)
  constexpr size_t nQ = (size_t)kBS * kAgents * kEmbed;      // 16M
  constexpr size_t nKV = (size_t)kBS * kNE * kEmbed;         // 64M
  constexpr size_t nE = (size_t)kBS * kNE * kInDim;          // 64M
  constexpr size_t nW = (size_t)3 * kEmbed * kInDim;         // 768K
  float* Qw = (float*)d_ws;
  float* Kw = Qw + nQ;
  float* Vw = Kw + nKV;
  _Float16* Eh = (_Float16*)(Vw + nKV);
  _Float16* El = Eh + nE;
  _Float16* Wh = El + nE;
  _Float16* Wl = Wh + nW;

  // pre-split f32 -> f16 hi/lo images (exact: x == hi + lo to f32 rounding)
  split_f16_kernel<<<(unsigned)(nE / 4 / 256), 256, 0, stream>>>(entities, Eh,
                                                                 El, nE / 4);
  split_f16_kernel<<<(unsigned)(nW / 4 / 256), 256, 0, stream>>>(W_in, Wh, Wl,
                                                                 nW / 4);

  // K|V projection: all 131072 entity rows, N = 1024 (W rows 512..1535)
  qkv_gemm_kernel<kNE><<<dim3(8, 1024), 256, 0, stream>>>(
      Eh, El, Wh + (size_t)kEmbed * kInDim, Wl + (size_t)kEmbed * kInDim, Kw,
      Vw);
  // Q projection: first 32 entities per batch (gathered rows), N = 512
  qkv_gemm_kernel<kAgents><<<dim3(4, 256), 256, 0, stream>>>(Eh, El, Wh, Wl, Qw,
                                                             Vw /*unused*/);
  // fused attention + output projection, one workgroup per batch
  attn_out_kernel<<<dim3(kBS), 256, 0, stream>>>(Qw, Kw, Vw, pre_mask,
                                                 post_mask, W_out, b_out, out);
}
